// AttentionBlock_87067577025419
// MI455X (gfx1250) — matrix operand order repaired and checked by reading, not compiled, not run on hardware
//
#include <hip/hip_runtime.h>

// ---------------------------------------------------------------------------
// MI455X (gfx1250) attention block: GroupNorm+SiLU -> QKV (bf16 WMMA GEMM) ->
// flash attention (bf16 WMMA, online softmax, TDM V-tile staging) ->
// out proj + residual (f32).
// B=4, C=512, H=W=64 (N=4096), GROUPS=32. wave32, v_wmma_f32_16x16x32_bf16.
// ---------------------------------------------------------------------------

typedef __attribute__((ext_vector_type(16))) __bf16 v16bf;
typedef __attribute__((ext_vector_type(8)))  __bf16 v8bf;
typedef __attribute__((ext_vector_type(4)))  __bf16 v4bf;
typedef __attribute__((ext_vector_type(8)))  float  v8f;
typedef __attribute__((ext_vector_type(4)))  unsigned int u32x4;
typedef __attribute__((ext_vector_type(8)))  int i32x8;
typedef __attribute__((ext_vector_type(4)))  int i32x4;

#define CDIM 512
#define NDIM 4096
#define BDIM 4
#define QK_SCALE 0.04419417382415922f   // 512^-0.5

#ifndef __has_builtin
#define __has_builtin(x) 0
#endif
#if __has_builtin(__builtin_amdgcn_tensor_load_to_lds) && \
    __has_builtin(__builtin_amdgcn_s_wait_tensorcnt)
#define USE_TDM 1
#else
#define USE_TDM 0
#endif

static __device__ __forceinline__ v8f wmma_bf16(v16bf a, v16bf b, v8f c) {
  return __builtin_amdgcn_wmma_f32_16x16x32_bf16(false, a, false, b,
                                                 (short)0, c, false, false);
}

// Build a 16-element bf16 fragment from two contiguous 8-element LDS chunks.
// off2 = element offset of the second chunk (16 for both A and B frags).
static __device__ __forceinline__ v16bf mk16(const __bf16* p, int off2) {
  v8bf lo = *(const v8bf*)p;
  v8bf hi = *(const v8bf*)(p + off2);
  return __builtin_shufflevector(lo, hi, 0,1,2,3,4,5,6,7,8,9,10,11,12,13,14,15);
}

// ---------------------------------------------------------------------------
// Kernel 1: f32 -> bf16 elementwise (weights)
// ---------------------------------------------------------------------------
__global__ __launch_bounds__(256) void f32_to_bf16(const float* __restrict__ in,
                                                   __bf16* __restrict__ out,
                                                   int n4) {
  int i = blockIdx.x * 256 + threadIdx.x;
  if (i < n4) {
    float4 v = ((const float4*)in)[i];
    v4bf o = {(__bf16)v.x, (__bf16)v.y, (__bf16)v.z, (__bf16)v.w};
    ((v4bf*)out)[i] = o;
  }
}

// ---------------------------------------------------------------------------
// Kernel 2: GroupNorm(32 groups) + SiLU, bf16 output.
// One workgroup per (b, g): 16 channels x 4096 = 65536 elements.
// ---------------------------------------------------------------------------
__global__ __launch_bounds__(256) void gn_silu(const float* __restrict__ x,
                                               const float* __restrict__ gamma,
                                               const float* __restrict__ beta,
                                               __bf16* __restrict__ xnb) {
  __shared__ float red0[256];
  __shared__ float red1[256];
  const int tid = threadIdx.x;
  const int b = blockIdx.x >> 5;
  const int g = blockIdx.x & 31;
  const int c0 = g * 16;
  const size_t base = ((size_t)b * CDIM + c0) * NDIM;
  const float4* xb = (const float4*)(x + base);

  float sum = 0.f, sq = 0.f;
  for (int i = tid; i < 16384; i += 256) {
    float4 v = xb[i];
    sum += v.x + v.y + v.z + v.w;
    sq  += v.x*v.x + v.y*v.y + v.z*v.z + v.w*v.w;
  }
  red0[tid] = sum; red1[tid] = sq;
  __syncthreads();
  for (int s = 128; s > 0; s >>= 1) {
    if (tid < s) { red0[tid] += red0[tid + s]; red1[tid] += red1[tid + s]; }
    __syncthreads();
  }
  const float inv_n = 1.f / 65536.f;
  const float mean = red0[0] * inv_n;
  const float var  = red1[0] * inv_n - mean * mean;
  const float rstd = rsqrtf(var + 1e-5f);

  v4bf* ob = (v4bf*)(xnb + base);
  for (int i = tid; i < 16384; i += 256) {
    float4 v = xb[i];
    int c = c0 + (i >> 10);           // 1024 float4 per channel
    float gm = gamma[c], bt = beta[c];
    float y0 = (v.x - mean) * rstd * gm + bt;
    float y1 = (v.y - mean) * rstd * gm + bt;
    float y2 = (v.z - mean) * rstd * gm + bt;
    float y3 = (v.w - mean) * rstd * gm + bt;
    y0 = y0 / (1.f + __expf(-y0));
    y1 = y1 / (1.f + __expf(-y1));
    y2 = y2 / (1.f + __expf(-y2));
    y3 = y3 / (1.f + __expf(-y3));
    v4bf o = {(__bf16)y0, (__bf16)y1, (__bf16)y2, (__bf16)y3};
    ob[i] = o;
  }
}

// ---------------------------------------------------------------------------
// Kernel 3: Y[b, m, n] = W[m, :] . X[b, :, n]  (M=K=512, N=4096 per batch)
// Tiles: 128x128x32, 8 waves in 2x4, each wave 64x32 (8 wmma accum tiles).
// F32OUT=false: Yb = (acc+bias)*outScale (bf16)
// F32OUT=true : Yf = acc + bias + resid  (f32)
// ---------------------------------------------------------------------------
#define AP 40   // padded K stride of LDS tiles
template <bool F32OUT>
__global__ __launch_bounds__(256) void gemm512(const __bf16* __restrict__ W,
                                               const __bf16* __restrict__ X,
                                               const float* __restrict__ bias,
                                               const float* __restrict__ resid,
                                               __bf16* __restrict__ Yb,
                                               float* __restrict__ Yf,
                                               float outScale) {
  __shared__ __bf16 Als[128 * AP];   // [m][k]
  __shared__ __bf16 Bls[128 * AP];   // [n][k] (transposed on fill)
  const int tid  = threadIdx.x;
  const int lane = tid & 31, wid = tid >> 5;
  const int nloc = lane & 15, hi = lane >> 4;
  const int kbA = hi ? 8 : 0, kbB = hi ? 8 : 0, msel = hi ? 8 : 0;
  const int m0 = blockIdx.y * 128;
  const int n0 = blockIdx.x * 128;
  const size_t xbase = (size_t)blockIdx.z * CDIM * NDIM;
  const int wm = wid >> 2, wn = wid & 3;      // 2 x 4 wave grid
  const int arow = tid >> 1, akc = (tid & 1) * 16;
  const int brow = tid >> 3, bnc = (tid & 7) * 16;

  v8f acc[4][2];
#pragma unroll
  for (int mt = 0; mt < 4; ++mt)
#pragma unroll
    for (int nt = 0; nt < 2; ++nt)
#pragma unroll
      for (int r = 0; r < 8; ++r) acc[mt][nt][r] = 0.f;

  for (int k0 = 0; k0 < CDIM; k0 += 32) {
    // A tile: W rows, k-contiguous -> direct
    {
      const v8bf* g = (const v8bf*)(W + (size_t)(m0 + arow) * CDIM + k0 + akc);
      *(v8bf*)(Als + arow * AP + akc)     = g[0];
      *(v8bf*)(Als + arow * AP + akc + 8) = g[1];
    }
    // B tile: X[k][n] -> transpose into Bls[n][k]
    {
      const v8bf* g = (const v8bf*)(X + xbase + (size_t)(k0 + brow) * NDIM + n0 + bnc);
      v8bf x0 = g[0], x1 = g[1];
#pragma unroll
      for (int e = 0; e < 8; ++e) {
        Bls[(bnc + e) * AP + brow]     = x0[e];
        Bls[(bnc + 8 + e) * AP + brow] = x1[e];
      }
    }
    __syncthreads();
    v16bf af[4], bfr[2];
#pragma unroll
    for (int mt = 0; mt < 4; ++mt)
      af[mt] = mk16(Als + (wm * 64 + mt * 16 + nloc) * AP + kbA, 16);
#pragma unroll
    for (int nt = 0; nt < 2; ++nt)
      bfr[nt] = mk16(Bls + (wn * 32 + nt * 16 + nloc) * AP + kbB, 16);
#pragma unroll
    for (int mt = 0; mt < 4; ++mt)
#pragma unroll
      for (int nt = 0; nt < 2; ++nt)
        acc[mt][nt] = wmma_bf16(af[mt], bfr[nt], acc[mt][nt]);
    __syncthreads();
  }

#pragma unroll
  for (int mt = 0; mt < 4; ++mt)
#pragma unroll
    for (int nt = 0; nt < 2; ++nt) {
      const int n = n0 + wn * 32 + nt * 16 + nloc;
#pragma unroll
      for (int r = 0; r < 8; ++r) {
        const int m = m0 + wm * 64 + mt * 16 + msel + r;
        const size_t idx = xbase + (size_t)m * NDIM + n;
        const float v = acc[mt][nt][r] + bias[m];
        if (F32OUT) Yf[idx] = v + resid[idx];
        else        Yb[idx] = (__bf16)(v * outScale);
      }
    }
}

// ---------------------------------------------------------------------------
// Kernel 4: flash attention. Per WG: 32 i-rows, all C=512; stream j in 64s.
// LDS: Qs[32][520] (i-major), Ks[64][520] (j-major), Vs[512][72] (c-major),
//      Ps[32][72], Ss[32][64] f32, row stats. ~182 KB of 320 KB/WGP.
// V tile is staged by the Tensor Data Mover (pad_enable reproduces the JP=72
// padded layout); K tile is transposed through VGPRs by all 8 waves.
// ---------------------------------------------------------------------------
#define CP 520
#define JP 72
#define SM_KS  33280
#define SM_VS  99840
#define SM_PS  173568
#define SM_SS  178176
#define SM_M   186368
#define SM_L   186496
#define SM_SC  186624
#define SM_TOT 186752

__global__ __launch_bounds__(256) void flash32(const __bf16* __restrict__ Q,
                                               const __bf16* __restrict__ K,
                                               const __bf16* __restrict__ V,
                                               __bf16* __restrict__ O) {
  extern __shared__ char smem[];
  __bf16* Qs = (__bf16*)smem;
  __bf16* Ks = (__bf16*)(smem + SM_KS);
  __bf16* Vs = (__bf16*)(smem + SM_VS);
  __bf16* Ps = (__bf16*)(smem + SM_PS);
  float*  Ss = (float*)(smem + SM_SS);
  float*  mrow = (float*)(smem + SM_M);
  float*  lrow = (float*)(smem + SM_L);
  float*  srow = (float*)(smem + SM_SC);

  const int tid  = threadIdx.x;
  const int lane = tid & 31, wid = tid >> 5;
  const int nloc = lane & 15, hi = lane >> 4;
  const int kbA = hi ? 8 : 0, kbB = hi ? 8 : 0, msel = hi ? 8 : 0;
  const int i0 = blockIdx.x * 32;
  const size_t base = (size_t)blockIdx.y * CDIM * NDIM;

  // Stage Q^T: Qs[i][c] from Q[c][i0+i]
  for (int idx = tid; idx < 1024; idx += 256) {
    const int c = idx >> 1, ic = (idx & 1) * 16;
    const v8bf* g = (const v8bf*)(Q + base + (size_t)c * NDIM + i0 + ic);
    v8bf q0 = g[0], q1 = g[1];
#pragma unroll
    for (int e = 0; e < 8; ++e) {
      Qs[(ic + e) * CP + c]     = q0[e];
      Qs[(ic + 8 + e) * CP + c] = q1[e];
    }
  }
  if (tid < 32) { mrow[tid] = -1e30f; lrow[tid] = 0.f; }

  v8f oacc[4][2];
#pragma unroll
  for (int mt = 0; mt < 4; ++mt)
#pragma unroll
    for (int nt = 0; nt < 2; ++nt)
#pragma unroll
      for (int r = 0; r < 8; ++r) oacc[mt][nt][r] = 0.f;
  __syncthreads();

  const int si = (wid & 1) * 16;   // S-phase i tile
  const int sj = (wid >> 1) * 16;  // S-phase j tile
  const int cb = wid * 64;         // O-phase channel base

  for (int j0 = 0; j0 < NDIM; j0 += 64) {
#if USE_TDM
    // --- TDM: async DMA of V tile [512 c][64 j] -> Vs[c][JP] (padded rows).
    // 2D tile, data_size=2B, pad 4 DWORDs (16B) after every 32 DWORDs (128B
    // = one 64-elem row): exactly the JP=72 layout. Issued once (wave 0).
    if (wid == 0) {
      const unsigned long long va =
          (unsigned long long)(const void*)(V + base + j0);
      u32x4 g0 = { 1u, (unsigned)SM_VS,
                   (unsigned)(va & 0xffffffffull),
                   (unsigned)((va >> 32) & 0x1ffffffull) | (2u << 30) };
      i32x8 g1 = { (int)((1u << 16) | (1u << 20) | (4u << 22) | (3u << 25)),
                   (int)(4096u << 16),   // tensor_dim0 = 4096 (bits 79:48)
                   (int)(512u << 16),    // tensor_dim1 = 512  (bits 111:80)
                   (int)(64u << 16),     // tile_dim0 = 64     (bits 127:112)
                   512,                  // tile_dim1 = 512, tile_dim2 = 0
                   4096,                 // tensor_dim0_stride = 4096
                   0, 0 };
      i32x4 gz = { 0, 0, 0, 0 };
#if defined(__clang_major__) && __clang_major__ >= 23
      i32x8 gz8 = { 0, 0, 0, 0, 0, 0, 0, 0 };
      __builtin_amdgcn_tensor_load_to_lds(g0, g1, gz, gz, gz8, 0);
#else
      __builtin_amdgcn_tensor_load_to_lds(g0, g1, gz, gz, 0);
#endif
    }
#endif
    // Stage K^T (Ks[j][c]) by all waves; prefetch next block.
    for (int idx = tid; idx < 2048; idx += 256) {
      const int c = idx >> 2, jc = (idx & 3) * 16;
      const size_t goff = base + (size_t)c * NDIM + j0 + jc;
      const v8bf* gk = (const v8bf*)(K + goff);
      v8bf k0v = gk[0], k1v = gk[1];
#pragma unroll
      for (int e = 0; e < 8; ++e) {
        Ks[(jc + e) * CP + c]     = k0v[e];
        Ks[(jc + 8 + e) * CP + c] = k1v[e];
      }
#if !USE_TDM
      const v8bf* gv = (const v8bf*)(V + goff);
      *(v8bf*)(Vs + c * JP + jc)     = gv[0];
      *(v8bf*)(Vs + c * JP + jc + 8) = gv[1];
#endif
      if (j0 + 64 < NDIM) {
        __builtin_prefetch((const void*)(K + goff + 64), 0, 1);
        __builtin_prefetch((const void*)(V + goff + 64), 0, 1);
      }
    }
#if USE_TDM
    if (wid == 0) __builtin_amdgcn_s_wait_tensorcnt(0);
#endif
    __syncthreads();

    // S = Q^T K : one 16x16 tile per wave, K-dim 512 (16 wmma)
    v8f s;
#pragma unroll
    for (int r = 0; r < 8; ++r) s[r] = 0.f;
#pragma unroll
    for (int ks = 0; ks < CDIM; ks += 32) {
      v16bf aq = mk16(Qs + (si + nloc) * CP + ks + kbA, 16);
      v16bf bk = mk16(Ks + (sj + nloc) * CP + ks + kbB, 16);
      s = wmma_bf16(aq, bk, s);
    }
#pragma unroll
    for (int r = 0; r < 8; ++r)
      Ss[(si + msel + r) * 64 + sj + nloc] = s[r];
    __syncthreads();

    // Online softmax: one thread per i-row, float4-vectorized.
    if (tid < 32) {
      const float4* srw = (const float4*)(Ss + tid * 64);
      float m = mrow[tid];
      float mx = m;
      for (int j = 0; j < 16; ++j) {
        float4 v = srw[j];
        mx = fmaxf(mx, fmaxf(fmaxf(v.x, v.y), fmaxf(v.z, v.w)));
      }
      const float sc = __expf(m - mx);
      float sum = 0.f;
      v4bf* prw = (v4bf*)(Ps + tid * JP);
      for (int j = 0; j < 16; ++j) {
        float4 v = srw[j];
        float p0 = __expf(v.x - mx), p1 = __expf(v.y - mx);
        float p2 = __expf(v.z - mx), p3 = __expf(v.w - mx);
        sum += (p0 + p1) + (p2 + p3);
        v4bf o = {(__bf16)p0, (__bf16)p1, (__bf16)p2, (__bf16)p3};
        prw[j] = o;
      }
      mrow[tid] = mx;
      lrow[tid] = lrow[tid] * sc + sum;
      srow[tid] = sc;
    }
    __syncthreads();

    // Rescale O and accumulate O += V . P^T (K-dim 64 -> 2 wmma per tile)
#pragma unroll
    for (int nt = 0; nt < 2; ++nt) {
      const float sc = srow[nt * 16 + nloc];
#pragma unroll
      for (int mt = 0; mt < 4; ++mt) oacc[mt][nt] *= sc;
    }
#pragma unroll
    for (int ks = 0; ks < 64; ks += 32) {
      v16bf bp[2];
#pragma unroll
      for (int nt = 0; nt < 2; ++nt)
        bp[nt] = mk16(Ps + (nt * 16 + nloc) * JP + ks + kbB, 16);
#pragma unroll
      for (int mt = 0; mt < 4; ++mt) {
        v16bf av = mk16(Vs + (cb + mt * 16 + nloc) * JP + ks + kbA, 16);
#pragma unroll
        for (int nt = 0; nt < 2; ++nt)
          oacc[mt][nt] = wmma_bf16(av, bp[nt], oacc[mt][nt]);
      }
    }
    __syncthreads();
  }

  // Epilogue: O /= l, store bf16 [c][i]
#pragma unroll
  for (int nt = 0; nt < 2; ++nt) {
    const float linv = 1.f / lrow[nt * 16 + nloc];
    const int i = i0 + nt * 16 + nloc;
#pragma unroll
    for (int mt = 0; mt < 4; ++mt)
#pragma unroll
      for (int r = 0; r < 8; ++r) {
        const int c = cb + mt * 16 + msel + r;
        O[base + (size_t)c * NDIM + i] = (__bf16)(oacc[mt][nt][r] * linv);
      }
  }
}

// ---------------------------------------------------------------------------
// Host launcher
// ---------------------------------------------------------------------------
extern "C" void kernel_launch(void* const* d_in, const int* in_sizes, int n_in,
                              void* d_out, int out_size, void* d_ws, size_t ws_size,
                              hipStream_t stream) {
  (void)in_sizes; (void)n_in; (void)out_size; (void)ws_size;
  const float* x     = (const float*)d_in[0];
  const float* gamma = (const float*)d_in[1];
  const float* beta  = (const float*)d_in[2];
  const float* wq = (const float*)d_in[3];  const float* bq = (const float*)d_in[4];
  const float* wk = (const float*)d_in[5];  const float* bk = (const float*)d_in[6];
  const float* wv = (const float*)d_in[7];  const float* bv = (const float*)d_in[8];
  const float* wo = (const float*)d_in[9];  const float* bo = (const float*)d_in[10];
  float* out = (float*)d_out;

  char* ws = (char*)d_ws;
  const size_t ACT = (size_t)BDIM * CDIM * NDIM * sizeof(__bf16);  // 16 MB
  const size_t WB  = (size_t)CDIM * CDIM * sizeof(__bf16);         // 0.5 MB
  __bf16* xnb = (__bf16*)(ws);
  __bf16* qb  = (__bf16*)(ws + 1 * ACT);
  __bf16* kb  = (__bf16*)(ws + 2 * ACT);
  __bf16* vb  = (__bf16*)(ws + 3 * ACT);
  __bf16* ob  = (__bf16*)(ws + 4 * ACT);
  __bf16* wqb = (__bf16*)(ws + 5 * ACT);
  __bf16* wkb = (__bf16*)(ws + 5 * ACT + 1 * WB);
  __bf16* wvb = (__bf16*)(ws + 5 * ACT + 2 * WB);
  __bf16* wob = (__bf16*)(ws + 5 * ACT + 3 * WB);

  // 1. Weights -> bf16
  const int n4 = CDIM * CDIM / 4;
  f32_to_bf16<<<256, 256, 0, stream>>>(wq, wqb, n4);
  f32_to_bf16<<<256, 256, 0, stream>>>(wk, wkb, n4);
  f32_to_bf16<<<256, 256, 0, stream>>>(wv, wvb, n4);
  f32_to_bf16<<<256, 256, 0, stream>>>(wo, wob, n4);

  // 2. GroupNorm + SiLU
  gn_silu<<<BDIM * 32, 256, 0, stream>>>(x, gamma, beta, xnb);

  // 3. QKV projections (scale C^-0.5 folded into Q)
  dim3 gg(NDIM / 128, CDIM / 128, BDIM);
  gemm512<false><<<gg, 256, 0, stream>>>(wqb, xnb, bq, nullptr, qb, nullptr, QK_SCALE);
  gemm512<false><<<gg, 256, 0, stream>>>(wkb, xnb, bk, nullptr, kb, nullptr, 1.0f);
  gemm512<false><<<gg, 256, 0, stream>>>(wvb, xnb, bv, nullptr, vb, nullptr, 1.0f);

  // 4. Flash attention
  dim3 fg(NDIM / 32, BDIM);
  flash32<<<fg, 256, SM_TOT, stream>>>(qb, kb, vb, ob);

  // 5. Output projection + bias + residual (f32)
  gemm512<true><<<gg, 256, 0, stream>>>(wob, ob, bo, x, nullptr, out, 1.0f);
}
